// ActorUpAction_80796924772661
// MI455X (gfx1250) — matrix operand order repaired and checked by reading, not compiled, not run on hardware
//
#include <hip/hip_runtime.h>
#include <math.h>

typedef _Float16 h16;
typedef __attribute__((ext_vector_type(16))) _Float16 v16h;
typedef __attribute__((ext_vector_type(8)))  _Float16 v8h;
typedef __attribute__((ext_vector_type(8)))  float    v8f;

#define THREADS 256
#define ROWS    128          // rows per workgroup (8 waves x 16 rows)

// ---- LDS layout (bytes). Total ~292 KB <= 320 KB per WGP (dynamic LDS). ----
#define OFF_W    0           // 131072 B : f16 weight staging (W1 -> W2 -> fc1/fc2/fc3)
#define OFF_HA   131072      //  65536 B : [128][256] f16  tanh(x)|tanh(m)
#define OFF_HB   196608      //  65536 B : [128][256] f16  h1 staging / msg scratch
#define OFF_XR   262144      //  32768 B : [128][128] f16  raw x (fc1 input)
#define OFF_NORM OFF_HB      // [128][64] f32 (fc1 pre-norm), later [128][32] f32 (fc3)
#define OFF_ZX   229376      // [128][64] f16 tanh(normalized hx)
#define OFF_Z2   245760      // [128][64] f16 tanh(fc2 out)
#define OFF_BIAS 294912      // f32: [0:256) b1 [256:512) b2 [512:768) w3
                             //      [768:832) fc1b [832:896) fc2b [896:928) fc3b [928] b3
#define LDS_BYTES 299008

// ---- Workspace layout for pre-converted f16 weights (halves) ----
#define WS_W1   0            // 256*256
#define WS_W2   65536        // 256*256
#define WS_FC1  131072       // 64*128
#define WS_FC2  139264       // 64*192
#define WS_FC3  151552       // 32*64
#define WS_HALVES 153600
#define WS_NEED_BYTES (WS_HALVES * 2)

// ---------------- WMMA helpers (CDNA5 16x16x32 f16 -> f32) ----------------
__device__ __forceinline__ v8f wmma_acc(v16h a, v16h b, v8f c) {
  // (neg_a, A, neg_b, B, c_mod, C, reuse_a, reuse_b)
  return __builtin_amdgcn_wmma_f32_16x16x32_f16(false, a, false, b, (short)0, c,
                                                false, false);
}

__device__ __forceinline__ v8f splat8(float v) {
  v8f r;
#pragma unroll
  for (int i = 0; i < 8; ++i) r[i] = v;
  return r;
}

// A fragment, 16x32 f16, row-major LDS source (ISA 7.12.2 16-bit A layout):
// lanes 0-15: M=lane, VGPR0-3 => K=k0..k0+7, VGPR4-7 => K=k0+16..k0+23 (k0 = kt*32)
// lanes 16-31: same M, K offset +8.
__device__ __forceinline__ v16h frag_ldA(const h16* __restrict__ buf, int stride,
                                         int row0, int kt, int lane) {
  const int r  = row0 + (lane & 15);
  const int k0 = kt * 32 + ((lane >> 4) << 3);
  const h16* p = buf + r * stride + k0;
  v8h lo = *(const v8h*)(p);
  v8h hi = *(const v8h*)(p + 16);
  v16h v;
#pragma unroll
  for (int i = 0; i < 8; ++i) { v[i] = lo[i]; v[i + 8] = hi[i]; }
  return v;
}

// B fragment, 32x16 f16. Same K order as A: lanes 0-15 hold K=k0..k0+7 and
// k0+16..k0+23, lanes 16-31 K offset +8 (k0 = kt*32); N = lane&15. Weight matrix is
// row-major [N][K] (torch Linear weight), so B[k][n] = W[n][k].
__device__ __forceinline__ v16h frag_ldB(const h16* __restrict__ buf, int stride,
                                         int n0, int kt, int lane) {
  const int n  = n0 + (lane & 15);
  const int k0 = kt * 32 + ((lane >> 4) << 3);
  const h16* p = buf + n * stride + k0;
  v8h lo = *(const v8h*)(p);
  v8h hi = *(const v8h*)(p + 16);
  v16h v;
#pragma unroll
  for (int i = 0; i < 8; ++i) { v[i] = lo[i]; v[i + 8] = hi[i]; }
  return v;
}

// Cooperative f32 -> f16 staging of a weight matrix into LDS (fallback path).
__device__ __forceinline__ void coop_w(const float* __restrict__ g,
                                       h16* __restrict__ l, int n, int tid) {
  for (int i = tid * 4; i < n; i += THREADS * 4) {
    const float4 v = *(const float4*)(g + i);
    l[i + 0] = (h16)v.x; l[i + 1] = (h16)v.y;
    l[i + 2] = (h16)v.z; l[i + 3] = (h16)v.w;
  }
}

// Cooperative f16 -> f16 LDS fill with b128 copies (fast path, pre-converted ws).
__device__ __forceinline__ void coop_cp(const h16* __restrict__ g,
                                        h16* __restrict__ l, int nhalf, int tid) {
  const uint4* gs = (const uint4*)g;
  uint4*       ls = (uint4*)l;
  const int nq = nhalf >> 3;
  for (int i = tid; i < nq; i += THREADS) ls[i] = gs[i];
}

template <bool PRE>
__global__ __launch_bounds__(THREADS, 1)
void actor_fused(const float* __restrict__ x,     const float* __restrict__ m,
                 const float* __restrict__ fc1_w, const float* __restrict__ fc1_b,
                 const float* __restrict__ fc2_w, const float* __restrict__ fc2_b,
                 const float* __restrict__ fc3_w, const float* __restrict__ fc3_b,
                 const float* __restrict__ ab_w1, const float* __restrict__ ab_b1,
                 const float* __restrict__ ab_w2, const float* __restrict__ ab_b2,
                 const float* __restrict__ ab_w3, const float* __restrict__ ab_b3,
                 const h16* __restrict__ wsw,
                 float* __restrict__ out_msg, float* __restrict__ out_act) {
  extern __shared__ __align__(16) char smem[];
  h16*   wbuf = (h16*)(smem + OFF_W);
  h16*   hA   = (h16*)(smem + OFF_HA);
  h16*   hB   = (h16*)(smem + OFF_HB);
  h16*   xr   = (h16*)(smem + OFF_XR);
  float* nbuf = (float*)(smem + OFF_NORM);
  h16*   zx   = (h16*)(smem + OFF_ZX);
  h16*   z2   = (h16*)(smem + OFF_Z2);
  float* lb   = (float*)(smem + OFF_BIAS);

  const int    tid  = threadIdx.x;
  const int    lane = tid & 31;
  const int    wv   = tid >> 5;
  const int    r0   = wv * 16;                 // this wave's 16 rows (local)
  const size_t g0   = (size_t)blockIdx.x * ROWS;

  // ---------------- Stage 0: W1 + biases + inputs into LDS ----------------
  if (PRE) coop_cp(wsw + WS_W1, wbuf, 256 * 256, tid);
  else     coop_w(ab_w1, wbuf, 256 * 256, tid);
  if (tid < 256) {
    lb[tid]       = ab_b1[tid];
    lb[256 + tid] = ab_b2[tid];
    lb[512 + tid] = ab_w3[tid];
  }
  if (tid < 64) {
    lb[768 + tid] = fc1_b[tid];
    lb[832 + tid] = fc2_b[tid];
  }
  if (tid < 32) lb[896 + tid] = fc3_b[tid];
  if (tid == 0) lb[928] = ab_b3[0];
  for (int i = tid * 4; i < ROWS * 128; i += THREADS * 4) {
    const int r = i >> 7, c = i & 127;
    const float4 v = *(const float4*)(x + (g0 + r) * 128 + c);
    hA[r * 256 + c + 0] = (h16)tanhf(v.x);
    hA[r * 256 + c + 1] = (h16)tanhf(v.y);
    hA[r * 256 + c + 2] = (h16)tanhf(v.z);
    hA[r * 256 + c + 3] = (h16)tanhf(v.w);
    xr[r * 128 + c + 0] = (h16)v.x;
    xr[r * 128 + c + 1] = (h16)v.y;
    xr[r * 128 + c + 2] = (h16)v.z;
    xr[r * 128 + c + 3] = (h16)v.w;
  }
  for (int i = tid * 4; i < ROWS * 128; i += THREADS * 4) {
    const int r = i >> 7, c = i & 127;
    const float4 v = *(const float4*)(m + (g0 + r) * 128 + c);
    hA[r * 256 + 128 + c + 0] = (h16)tanhf(v.x);
    hA[r * 256 + 128 + c + 1] = (h16)tanhf(v.y);
    hA[r * 256 + 128 + c + 2] = (h16)tanhf(v.z);
    hA[r * 256 + 128 + c + 3] = (h16)tanhf(v.w);
  }
  __syncthreads();

  // ---------------- Layer 1: h1 = relu(xm @ W1^T + b1) ----------------
  {
    v16h af[8];
#pragma unroll
    for (int kt = 0; kt < 8; ++kt) af[kt] = frag_ldA(hA, 256, r0, kt, lane);
    for (int nt = 0; nt < 16; ++nt) {
      const int n = nt * 16 + (lane & 15);
      v8f acc = splat8(lb[n]);
#pragma unroll
      for (int kt = 0; kt < 8; ++kt)
        acc = wmma_acc(af[kt], frag_ldB(wbuf, 256, nt * 16, kt, lane), acc);
      const int mb = r0 + ((lane >> 4) << 3);
#pragma unroll
      for (int r = 0; r < 8; ++r)
        hB[(mb + r) * 256 + n] = (h16)fmaxf(acc[r], 0.0f);
    }
  }
  __syncthreads();

  // ---------------- Swap W1 -> W2 ----------------
  if (PRE) coop_cp(wsw + WS_W2, wbuf, 256 * 256, tid);
  else     coop_w(ab_w2, wbuf, 256 * 256, tid);
  __syncthreads();

  // ---- Layer 2 + fused action head: action = tanh(relu(h1@W2^T+b2)@w3 + b3) ----
  {
    v16h af[8];
#pragma unroll
    for (int kt = 0; kt < 8; ++kt) af[kt] = frag_ldA(hB, 256, r0, kt, lane);
    v8f actsum = splat8(0.0f);
    for (int nt = 0; nt < 16; ++nt) {
      const int n = nt * 16 + (lane & 15);
      v8f acc = splat8(lb[256 + n]);
#pragma unroll
      for (int kt = 0; kt < 8; ++kt)
        acc = wmma_acc(af[kt], frag_ldB(wbuf, 256, nt * 16, kt, lane), acc);
      const float w3v = lb[512 + n];
#pragma unroll
      for (int r = 0; r < 8; ++r) actsum[r] += fmaxf(acc[r], 0.0f) * w3v;
    }
    // reduce over the 16 lanes of each half-wave (N dimension)
#pragma unroll
    for (int s = 1; s < 16; s <<= 1) {
#pragma unroll
      for (int r = 0; r < 8; ++r) actsum[r] += __shfl_xor(actsum[r], s, 32);
    }
    if ((lane & 15) == 0) {
      const int mb = r0 + ((lane >> 4) << 3);
      const float b3 = lb[928];
#pragma unroll
      for (int r = 0; r < 8; ++r)
        out_act[g0 + mb + r] = tanhf(actsum[r] + b3);
    }
  }
  __syncthreads();

  // ---------------- Stage fc1/fc2/fc3 weights ----------------
  h16* fc1buf = wbuf;                 // [64][128]
  h16* fc2buf = wbuf + 8192;          // [64][192]
  h16* fc3buf = wbuf + 20480;         // [32][64]
  if (PRE) {
    coop_cp(wsw + WS_FC1, fc1buf, 64 * 128 + 64 * 192 + 32 * 64, tid);  // contiguous
  } else {
    coop_w(fc1_w, fc1buf, 64 * 128, tid);
    coop_w(fc2_w, fc2buf, 64 * 192, tid);
    coop_w(fc3_w, fc3buf, 32 * 64, tid);
  }
  __syncthreads();

  // ---------------- fc1: pre = x @ fc1_w^T + b ----------------
  {
    v16h af[4];
#pragma unroll
    for (int kt = 0; kt < 4; ++kt) af[kt] = frag_ldA(xr, 128, r0, kt, lane);
    for (int nt = 0; nt < 4; ++nt) {
      const int n = nt * 16 + (lane & 15);
      v8f acc = splat8(lb[768 + n]);
#pragma unroll
      for (int kt = 0; kt < 4; ++kt)
        acc = wmma_acc(af[kt], frag_ldB(fc1buf, 128, nt * 16, kt, lane), acc);
      const int mb = r0 + ((lane >> 4) << 3);
#pragma unroll
      for (int r = 0; r < 8; ++r) nbuf[(mb + r) * 64 + n] = acc[r];
    }
  }
  __syncthreads();

  // l2norm + tanh -> zx   (lane&15 = row, lane>>4 picks 32-col half)
  {
    const int row = r0 + (lane & 15);
    const int c0  = (lane >> 4) * 32;
    float ss = 0.0f;
#pragma unroll
    for (int c = 0; c < 32; ++c) {
      const float v = nbuf[row * 64 + c0 + c];
      ss += v * v;
    }
    ss += __shfl_xor(ss, 16, 32);
    const float sc = 1.0f / fmaxf(sqrtf(ss), 1e-12f);
#pragma unroll
    for (int c = 0; c < 32; ++c)
      zx[row * 64 + c0 + c] = (h16)tanhf(nbuf[row * 64 + c0 + c] * sc);
  }
  __syncthreads();

  // ---------------- fc2: z2 = tanh(concat(zx, tanh(m)) @ fc2_w^T + b) --------
  {
    v16h af[6];
#pragma unroll
    for (int kt = 0; kt < 6; ++kt)
      af[kt] = (kt < 2) ? frag_ldA(zx, 64, r0, kt, lane)
                        : frag_ldA(hA + 128, 256, r0, kt - 2, lane);
    for (int nt = 0; nt < 4; ++nt) {
      const int n = nt * 16 + (lane & 15);
      v8f acc = splat8(lb[832 + n]);
#pragma unroll
      for (int kt = 0; kt < 6; ++kt)
        acc = wmma_acc(af[kt], frag_ldB(fc2buf, 192, nt * 16, kt, lane), acc);
      const int mb = r0 + ((lane >> 4) << 3);
#pragma unroll
      for (int r = 0; r < 8; ++r) z2[(mb + r) * 64 + n] = (h16)tanhf(acc[r]);
    }
  }
  __syncthreads();

  // ---------------- fc3: pre = z2 @ fc3_w^T + b ----------------
  {
    v16h af[2];
#pragma unroll
    for (int kt = 0; kt < 2; ++kt) af[kt] = frag_ldA(z2, 64, r0, kt, lane);
    for (int nt = 0; nt < 2; ++nt) {
      const int n = nt * 16 + (lane & 15);
      v8f acc = splat8(lb[896 + n]);
#pragma unroll
      for (int kt = 0; kt < 2; ++kt)
        acc = wmma_acc(af[kt], frag_ldB(fc3buf, 64, nt * 16, kt, lane), acc);
      const int mb = r0 + ((lane >> 4) << 3);
#pragma unroll
      for (int r = 0; r < 8; ++r) nbuf[(mb + r) * 32 + n] = acc[r];
    }
  }
  __syncthreads();

  // l2norm over 32 outputs -> global msg store
  {
    const int row = r0 + (lane & 15);
    const int c0  = (lane >> 4) * 16;
    float ss = 0.0f;
#pragma unroll
    for (int c = 0; c < 16; ++c) {
      const float v = nbuf[row * 32 + c0 + c];
      ss += v * v;
    }
    ss += __shfl_xor(ss, 16, 32);
    const float sc = 1.0f / fmaxf(sqrtf(ss), 1e-12f);
#pragma unroll
    for (int c = 0; c < 16; ++c)
      out_msg[(g0 + row) * 32 + c0 + c] = nbuf[row * 32 + c0 + c] * sc;
  }
}

// One-time fp32 -> f16 weight conversion into workspace.
__global__ void cvt_f16(const float* __restrict__ s, h16* __restrict__ d, int n) {
  for (int i = blockIdx.x * blockDim.x + threadIdx.x; i < n;
       i += gridDim.x * blockDim.x)
    d[i] = (h16)s[i];
}

extern "C" void kernel_launch(void* const* d_in, const int* in_sizes, int n_in,
                              void* d_out, int out_size, void* d_ws, size_t ws_size,
                              hipStream_t stream) {
  (void)n_in; (void)out_size;
  const float* x     = (const float*)d_in[0];
  const float* m     = (const float*)d_in[1];
  const float* fc1_w = (const float*)d_in[2];
  const float* fc1_b = (const float*)d_in[3];
  const float* fc2_w = (const float*)d_in[4];
  const float* fc2_b = (const float*)d_in[5];
  const float* fc3_w = (const float*)d_in[6];
  const float* fc3_b = (const float*)d_in[7];
  const float* ab_w1 = (const float*)d_in[8];
  const float* ab_b1 = (const float*)d_in[9];
  const float* ab_w2 = (const float*)d_in[10];
  const float* ab_b2 = (const float*)d_in[11];
  const float* ab_w3 = (const float*)d_in[12];
  const float* ab_b3 = (const float*)d_in[13];

  const int Btot = in_sizes[0] / 128;      // 262144
  float* out_msg = (float*)d_out;          // [B,32]
  float* out_act = out_msg + (size_t)Btot * 32;  // [B,1]
  const int nblocks = Btot / ROWS;         // 2048

  if (ws_size >= (size_t)WS_NEED_BYTES) {
    h16* ws = (h16*)d_ws;
    cvt_f16<<<128, 256, 0, stream>>>(ab_w1, ws + WS_W1, 256 * 256);
    cvt_f16<<<128, 256, 0, stream>>>(ab_w2, ws + WS_W2, 256 * 256);
    cvt_f16<<<32, 256, 0, stream>>>(fc1_w, ws + WS_FC1, 64 * 128);
    cvt_f16<<<32, 256, 0, stream>>>(fc2_w, ws + WS_FC2, 64 * 192);
    cvt_f16<<<8, 256, 0, stream>>>(fc3_w, ws + WS_FC3, 32 * 64);

    (void)hipFuncSetAttribute(reinterpret_cast<const void*>(actor_fused<true>),
                              hipFuncAttributeMaxDynamicSharedMemorySize,
                              LDS_BYTES);
    actor_fused<true><<<nblocks, THREADS, LDS_BYTES, stream>>>(
        x, m, fc1_w, fc1_b, fc2_w, fc2_b, fc3_w, fc3_b,
        ab_w1, ab_b1, ab_w2, ab_b2, ab_w3, ab_b3, ws, out_msg, out_act);
  } else {
    (void)hipFuncSetAttribute(reinterpret_cast<const void*>(actor_fused<false>),
                              hipFuncAttributeMaxDynamicSharedMemorySize,
                              LDS_BYTES);
    actor_fused<false><<<nblocks, THREADS, LDS_BYTES, stream>>>(
        x, m, fc1_w, fc1_b, fc2_w, fc2_b, fc3_w, fc3_b,
        ab_w1, ab_b1, ab_w2, ab_b2, ab_w3, ab_b3, (const h16*)nullptr,
        out_msg, out_act);
  }
}
